// DiG_generator_33440615366946
// MI455X (gfx1250) — matrix operand order repaired and checked by reading, not compiled, not run on hardware
//
#include <hip/hip_runtime.h>
#include <math.h>

// ---------------------------------------------------------------------------
// Problem constants (from reference): B=2, N=256, D=64, DD=768, H=12, DH=64,
// FF=2048, rows = B*N = 512 tokens for all encoder linears.
// ---------------------------------------------------------------------------
#define EPSF 1e-5f

typedef __attribute__((ext_vector_type(16))) __bf16 v16bf;
typedef __attribute__((ext_vector_type(8)))  float  v8f;

__device__ inline unsigned short f2bf(float f) {
    unsigned int u = __float_as_uint(f);
    unsigned int r = u + 0x7FFFu + ((u >> 16) & 1u);   // round-to-nearest-even
    return (unsigned short)(r >> 16);
}

__device__ inline unsigned int pack2bf(float a, float b) {
    return (unsigned int)f2bf(a) | ((unsigned int)f2bf(b) << 16);
}

__device__ inline v8f zero8() {
    v8f z;
    #pragma unroll
    for (int i = 0; i < 8; ++i) z[i] = 0.0f;
    return z;
}

// A-fragment (16x32 bf16, M x K) per CDNA5 ISA layout:
// lane<16: row M=lane, VGPR p holds K = (p>>2)*16 + (p&3)*2 (+1)
// lane>=16: row M=lane-16, same + 8.  Each VGPR = 2 consecutive bf16 = one u32.
__device__ inline v16bf load_frag_a(const unsigned short* base, int stride) {
    int lane = threadIdx.x & 31;
    int m = lane & 15, hi = lane >> 4;
    union { unsigned int u[8]; v16bf v; } f;
    const unsigned short* row = base + m * stride;
    #pragma unroll
    for (int p = 0; p < 8; ++p) {
        int k = ((p >> 2) << 4) + (hi << 3) + ((p & 3) << 1);
        f.u[p] = *reinterpret_cast<const unsigned int*>(row + k);
    }
    return f.v;
}

// B-fragment (32x16 bf16, K x N), staged n-major in LDS: lane holds column
// n = lane&15; VGPR p holds K = (p>>2)*16 + (p&3)*2 (+1), lanes>=16 same + 8.
__device__ inline v16bf load_frag_b(const unsigned short* base, int stride) {
    int lane = threadIdx.x & 31;
    int n = lane & 15, hi = lane >> 4;
    union { unsigned int u[8]; v16bf v; } f;
    const unsigned short* row = base + n * stride;
    #pragma unroll
    for (int p = 0; p < 8; ++p)
        f.u[p] = *reinterpret_cast<const unsigned int*>(row + ((p >> 2) << 4) + (hi << 3) + ((p & 3) << 1));
    return f.v;
}

// ---------------------------------------------------------------------------
// Generic WMMA GEMM: C(MxN) = A(MxK) * W(KxN) [+bias][epilogue]
// Tile: 128(M) x 64(N) per workgroup, 8 waves, each wave = 16 rows x 64 cols.
// ep: 0 = bias only, 1 = bias+relu, 2 = bias -> batchnorm -> relu
// Fast staging path (float4 global loads) when the tile is full and K is a
// multiple of 32; guarded scalar path only for the K=66 tail case.
// ---------------------------------------------------------------------------
__global__ __launch_bounds__(256) void k_gemm_bf16(
    const float* __restrict__ A, const float* __restrict__ W,
    const float* __restrict__ bias, float* __restrict__ C,
    int M, int N, int K, int ep,
    const float* __restrict__ bng, const float* __restrict__ bnb,
    const float* __restrict__ bnm, const float* __restrict__ bnv)
{
    __shared__ unsigned short As[128 * 36];   // 128 rows x 32 K (+4 pad)
    __shared__ unsigned short Bs[64 * 40];    // 64 cols  x 32 K (+8 pad), n-major

    const int tid = threadIdx.x;
    const int wv = tid >> 5, lane = tid & 31;
    const int n16 = lane & 15, hi = lane >> 4;
    const int mBase = blockIdx.y * 128;
    const int nBase = blockIdx.x * 64;
    const bool fast = ((K & 31) == 0) && (mBase + 128 <= M) && (nBase + 64 <= N);

    v8f acc[4];
    #pragma unroll
    for (int s = 0; s < 4; ++s) acc[s] = zero8();

    for (int k0 = 0; k0 < K; k0 += 32) {
        if (fast) {
            // A tile: thread -> row r = tid>>1, 16 contiguous cols, 4x float4
            {
                int r = tid >> 1, c0 = (tid & 1) << 4;
                const float4* src = reinterpret_cast<const float4*>(
                    A + (size_t)(mBase + r) * K + k0 + c0);
                unsigned short* dst = As + r * 36 + c0;
                #pragma unroll
                for (int q = 0; q < 4; ++q) {
                    float4 f = src[q];
                    unsigned int* d32 = reinterpret_cast<unsigned int*>(dst + (q << 2));
                    d32[0] = pack2bf(f.x, f.y);
                    d32[1] = pack2bf(f.z, f.w);
                }
            }
            // B tile: float4 along n, scatter 4 bf16 to n-major LDS
            #pragma unroll
            for (int q = 0; q < 2; ++q) {
                int lin = (tid << 1) + q;           // 0..511
                int kk = lin & 31, ng = lin >> 5;   // ng 0..15 (4 cols each)
                float4 f = *reinterpret_cast<const float4*>(
                    W + (size_t)(k0 + kk) * N + nBase + (ng << 2));
                int nb = ng << 2;
                Bs[(nb + 0) * 40 + kk] = f2bf(f.x);
                Bs[(nb + 1) * 40 + kk] = f2bf(f.y);
                Bs[(nb + 2) * 40 + kk] = f2bf(f.z);
                Bs[(nb + 3) * 40 + kk] = f2bf(f.w);
            }
            // prefetch next K tile into cache (global_prefetch_b8)
            if (k0 + 32 < K) {
                __builtin_prefetch(A + (size_t)(mBase + (tid >> 1)) * K + k0 + 32
                                     + ((tid & 1) << 4), 0, 3);
                __builtin_prefetch(W + (size_t)(k0 + 32 + (tid & 31)) * N + nBase
                                     + ((tid >> 5) << 3), 0, 3);
            }
        } else {
            // guarded scalar path (K tail / ragged tiles)
            for (int idx = tid; idx < 128 * 32; idx += 256) {
                int r = idx >> 5, c = idx & 31;
                int gm = mBase + r, gk = k0 + c;
                float v = (gm < M && gk < K) ? A[(size_t)gm * K + gk] : 0.0f;
                As[r * 36 + c] = f2bf(v);
            }
            for (int idx = tid; idx < 64 * 32; idx += 256) {
                int nn = idx >> 5, kk = idx & 31;
                int gn = nBase + nn, gk = k0 + kk;
                float v = (gn < N && gk < K) ? W[(size_t)gk * N + gn] : 0.0f;
                Bs[nn * 40 + kk] = f2bf(v);
            }
        }
        __syncthreads();

        v16bf a = load_frag_a(As + (wv << 4) * 36, 36);
        v16bf bfr[4];
        #pragma unroll
        for (int s = 0; s < 4; ++s)
            bfr[s] = load_frag_b(Bs + (s << 4) * 40, 40);
        #pragma unroll
        for (int s = 0; s < 4; ++s)
            acc[s] = __builtin_amdgcn_wmma_f32_16x16x32_bf16(
                false, a, false, bfr[s], (short)0, acc[s], false, false);
        __syncthreads();
    }

    // epilogue
    #pragma unroll
    for (int s = 0; s < 4; ++s) {
        int n = nBase + (s << 4) + n16;
        if (n >= N) continue;
        float bv = bias ? bias[n] : 0.0f;
        float sc = 1.0f, sh = 0.0f;
        if (ep == 2) {
            sc = bng[n] * rsqrtf(bnv[n] + EPSF);
            sh = bnb[n] - bnm[n] * sc;
        }
        #pragma unroll
        for (int r = 0; r < 8; ++r) {
            int m = mBase + (wv << 4) + r + (hi << 3);
            if (m >= M) continue;
            float v = acc[s][r] + bv;
            if (ep == 2) v = v * sc + sh;
            if (ep >= 1) v = fmaxf(v, 0.0f);
            C[(size_t)m * N + n] = v;
        }
    }
}

// ---------------------------------------------------------------------------
// Concat [visual(512x64) | detected(512x2)] -> (512x66)
// ---------------------------------------------------------------------------
__global__ void k_concat(const float* __restrict__ vis,
                         const float* __restrict__ det,
                         float* __restrict__ out)
{
    int idx = blockIdx.x * 256 + threadIdx.x;
    if (idx >= 512 * 66) return;
    int row = idx / 66, c = idx - row * 66;
    out[idx] = (c < 64) ? vis[row * 64 + c] : det[row * 2 + (c - 64)];
}

// ---------------------------------------------------------------------------
// Attention with S=2 sequence (torch batch_first=False semantics).
// qkv: (512 rows = s*256+bt, 2304).  One thread per (bt, head).
// ---------------------------------------------------------------------------
__global__ void k_attn(const float* __restrict__ qkv, float* __restrict__ o)
{
    int t = blockIdx.x * blockDim.x + threadIdx.x;
    if (t >= 256 * 12) return;
    int bt = t / 12, h = t - bt * 12;
    const float* q0 = qkv + (size_t)bt * 2304 + h * 64;
    const float* q1 = qkv + (size_t)(256 + bt) * 2304 + h * 64;
    const float* k0 = q0 + 768, *k1 = q1 + 768;
    const float* v0 = q0 + 1536, *v1 = q1 + 1536;

    float s00 = 0, s01 = 0, s10 = 0, s11 = 0;
    for (int d = 0; d < 64; ++d) {
        float a0 = q0[d], a1 = q1[d], b0 = k0[d], b1 = k1[d];
        s00 += a0 * b0; s01 += a0 * b1; s10 += a1 * b0; s11 += a1 * b1;
    }
    const float sc = 0.125f;  // 1/sqrt(64)
    s00 *= sc; s01 *= sc; s10 *= sc; s11 *= sc;
    float m0 = fmaxf(s00, s01), m1 = fmaxf(s10, s11);
    float e00 = __expf(s00 - m0), e01 = __expf(s01 - m0);
    float e10 = __expf(s10 - m1), e11 = __expf(s11 - m1);
    float r0 = 1.0f / (e00 + e01), r1 = 1.0f / (e10 + e11);
    float w00 = e00 * r0, w01 = e01 * r0, w10 = e10 * r1, w11 = e11 * r1;

    float* o0 = o + (size_t)bt * 768 + h * 64;
    float* o1 = o + (size_t)(256 + bt) * 768 + h * 64;
    for (int d = 0; d < 64; ++d) {
        float a = v0[d], b = v1[d];
        o0[d] = w00 * a + w01 * b;
        o1[d] = w10 * a + w11 * b;
    }
}

// ---------------------------------------------------------------------------
// x = LayerNorm(x + r) * g + b, in place. One 256-thread block per row (D=768).
// ---------------------------------------------------------------------------
__global__ __launch_bounds__(256) void k_add_ln(
    float* __restrict__ x, const float* __restrict__ r,
    const float* __restrict__ g, const float* __restrict__ b)
{
    const int D = 768;
    int row = blockIdx.x, tid = threadIdx.x;
    __shared__ float red[256];
    float vals[3], s = 0.0f;
    #pragma unroll
    for (int i = 0; i < 3; ++i) {
        int c = tid + i * 256;
        vals[i] = x[(size_t)row * D + c] + r[(size_t)row * D + c];
        s += vals[i];
    }
    red[tid] = s; __syncthreads();
    for (int off = 128; off; off >>= 1) {
        if (tid < off) red[tid] += red[tid + off];
        __syncthreads();
    }
    float mean = red[0] / D; __syncthreads();
    float sq = 0.0f;
    #pragma unroll
    for (int i = 0; i < 3; ++i) { float d = vals[i] - mean; sq += d * d; }
    red[tid] = sq; __syncthreads();
    for (int off = 128; off; off >>= 1) {
        if (tid < off) red[tid] += red[tid + off];
        __syncthreads();
    }
    float inv = rsqrtf(red[0] / D + EPSF);
    #pragma unroll
    for (int i = 0; i < 3; ++i) {
        int c = tid + i * 256;
        x[(size_t)row * D + c] = (vals[i] - mean) * inv * g[c] + b[c];
    }
}

// ---------------------------------------------------------------------------
// Fused pairwise score head. Per workgroup: 8(i) x 16(j) = 128 pairs.
//   a1 = relu(bn1(xi[i] + xj[j]))        (b1 pre-folded into xj)    256 ch
//   a2 = relu(bn2(a1 @ w2 + b2))                                    128 ch
//   a3 = relu(bn3(a2 @ w3 + b3))                                     64 ch
//   out[b,i,j] = a3 . w4 + b4
// w2/w3 resident in LDS (bf16, n-major); all GEMMs via WMMA bf16 f32-acc.
// LDS regions time-multiplexed: W2 -> A2, A1 -> A3.
// ---------------------------------------------------------------------------
#define PH_SK1 264   // A1 stride (halves): 256 + 8 pad
#define PH_SK2 264   // W2 stride (halves)
#define PH_SK3 136   // W3 stride (halves): 128 + 8 pad
#define PH_SA2 136   // A2 stride (halves)
#define PH_SA3 68    // A3 stride (floats): 64 + 4 pad
#define PH_OFF_W2  0u
#define PH_OFF_A1  67584u
#define PH_OFF_W3  135168u
#define PH_OFF_XI  152576u
#define PH_OFF_PAR 177152u
#define PH_SMEM    181248u

__global__ __launch_bounds__(256) void k_pairhead(
    const float* __restrict__ xi, const float* __restrict__ xj,
    const float* __restrict__ bn1g, const float* __restrict__ bn1b,
    const float* __restrict__ bn1m, const float* __restrict__ bn1v,
    const float* __restrict__ w2, const float* __restrict__ b2,
    const float* __restrict__ bn2g, const float* __restrict__ bn2b,
    const float* __restrict__ bn2m, const float* __restrict__ bn2v,
    const float* __restrict__ w3, const float* __restrict__ b3,
    const float* __restrict__ bn3g, const float* __restrict__ bn3b,
    const float* __restrict__ bn3m, const float* __restrict__ bn3v,
    const float* __restrict__ w4, const float* __restrict__ b4,
    float* __restrict__ out)
{
    extern __shared__ __align__(16) char smem[];
    unsigned short* W2s = (unsigned short*)(smem + PH_OFF_W2);
    unsigned short* A1  = (unsigned short*)(smem + PH_OFF_A1);
    unsigned short* W3s = (unsigned short*)(smem + PH_OFF_W3);
    float* xis = (float*)(smem + PH_OFF_XI);       // [8][256]
    float* xjs = xis + 8 * 256;                    // [16][256]
    float* par = (float*)(smem + PH_OFF_PAR);
    float* s1 = par;        float* t1 = s1 + 256;
    float* s2 = t1 + 256;   float* t2 = s2 + 128;
    float* s3 = t2 + 128;   float* t3 = s3 + 64;
    float* w4s = t3 + 64;                          // [64]
    unsigned short* A2 = W2s;                      // alias (after GEMM1)
    float* A3 = (float*)A1;                        // alias (after GEMM2)

    const int tid = threadIdx.x;
    const int wv = tid >> 5, lane = tid & 31;
    const int n16 = lane & 15, hi = lane >> 4;
    const int bB = blockIdx.z;
    const int i0 = blockIdx.y << 3;
    const int j0 = blockIdx.x << 4;

    // --- stage folded BN params: y = acc*s + t (bias folded into t) ---
    if (tid < 256) {
        float sv = bn1g[tid] * rsqrtf(bn1v[tid] + EPSF);
        s1[tid] = sv; t1[tid] = bn1b[tid] - bn1m[tid] * sv;
    }
    if (tid < 128) {
        float sv = bn2g[tid] * rsqrtf(bn2v[tid] + EPSF);
        s2[tid] = sv; t2[tid] = bn2b[tid] - bn2m[tid] * sv + b2[tid] * sv;
    }
    if (tid < 64) {
        float sv = bn3g[tid] * rsqrtf(bn3v[tid] + EPSF);
        s3[tid] = sv; t3[tid] = bn3b[tid] - bn3m[tid] * sv + b3[tid] * sv;
        w4s[tid] = w4[tid];
    }
    // --- stage weights (float4 loads, fp32 -> bf16, n-major scatter) ---
    for (int lin = tid; lin < 256 * 32; lin += 256) {   // w2 (256,128): 8192 f4
        int k = lin >> 5, ng = lin & 31;
        float4 f = *reinterpret_cast<const float4*>(w2 + ((size_t)k << 7) + (ng << 2));
        int nb = ng << 2;
        W2s[(nb + 0) * PH_SK2 + k] = f2bf(f.x);
        W2s[(nb + 1) * PH_SK2 + k] = f2bf(f.y);
        W2s[(nb + 2) * PH_SK2 + k] = f2bf(f.z);
        W2s[(nb + 3) * PH_SK2 + k] = f2bf(f.w);
    }
    for (int lin = tid; lin < 128 * 16; lin += 256) {   // w3 (128,64): 2048 f4
        int k = lin >> 4, ng = lin & 15;
        float4 f = *reinterpret_cast<const float4*>(w3 + ((size_t)k << 6) + (ng << 2));
        int nb = ng << 2;
        W3s[(nb + 0) * PH_SK3 + k] = f2bf(f.x);
        W3s[(nb + 1) * PH_SK3 + k] = f2bf(f.y);
        W3s[(nb + 2) * PH_SK3 + k] = f2bf(f.z);
        W3s[(nb + 3) * PH_SK3 + k] = f2bf(f.w);
    }
    // --- stage xi (8 rows) / xj (16 rows), float4 copies ---
    for (int lin = tid; lin < 8 * 64; lin += 256) {
        int r = lin >> 6, c4 = lin & 63;
        reinterpret_cast<float4*>(xis + r * 256)[c4] =
            reinterpret_cast<const float4*>(xi + ((size_t)(bB * 256 + i0 + r)) * 256)[c4];
    }
    for (int lin = tid; lin < 16 * 64; lin += 256) {
        int r = lin >> 6, c4 = lin & 63;
        reinterpret_cast<float4*>(xjs + r * 256)[c4] =
            reinterpret_cast<const float4*>(xj + ((size_t)(bB * 256 + j0 + r)) * 256)[c4];
    }
    __syncthreads();

    // --- layer 1: build A1 (128 pairs x 256 ch), 2 bf16 per b32 store ---
    {
        int p = tid >> 1;                  // pair 0..127
        int pi = p >> 4, pj = p & 15;
        const float* xr = xis + pi * 256;
        const float* yr = xjs + pj * 256;
        int c0 = (tid & 1) << 7;
        unsigned int* dst = reinterpret_cast<unsigned int*>(A1 + p * PH_SK1 + c0);
        #pragma unroll 8
        for (int c = 0; c < 128; c += 2) {
            int cc = c0 + c;
            float v0 = fmaxf((xr[cc] + yr[cc]) * s1[cc] + t1[cc], 0.0f);
            float v1 = fmaxf((xr[cc + 1] + yr[cc + 1]) * s1[cc + 1] + t1[cc + 1], 0.0f);
            dst[c >> 1] = pack2bf(v0, v1);
        }
    }
    __syncthreads();

    // --- GEMM1: C1(128x128) = A1(128x256) @ W2(256x128) ---
    v8f acc1[8];
    #pragma unroll
    for (int s = 0; s < 8; ++s) acc1[s] = zero8();
    for (int k0 = 0; k0 < 256; k0 += 32) {
        v16bf a = load_frag_a(A1 + (wv << 4) * PH_SK1 + k0, PH_SK1);
        v16bf bfr[8];
        #pragma unroll
        for (int s = 0; s < 8; ++s)
            bfr[s] = load_frag_b(W2s + (s << 4) * PH_SK2 + k0, PH_SK2);
        #pragma unroll
        for (int s = 0; s < 8; ++s)
            acc1[s] = __builtin_amdgcn_wmma_f32_16x16x32_bf16(
                false, a, false, bfr[s], (short)0, acc1[s], false, false);
    }
    __syncthreads();
    // epilogue 1 -> A2 (bf16, reuses W2 region)
    #pragma unroll
    for (int s = 0; s < 8; ++s) {
        int n = (s << 4) + n16;
        float sc = s2[n], sh = t2[n];
        #pragma unroll
        for (int r = 0; r < 8; ++r) {
            int m = (wv << 4) + r + (hi << 3);
            float v = fmaxf(acc1[s][r] * sc + sh, 0.0f);
            A2[m * PH_SA2 + n] = f2bf(v);
        }
    }
    __syncthreads();

    // --- GEMM2: C2(128x64) = A2(128x128) @ W3(128x64) ---
    v8f acc2[4];
    #pragma unroll
    for (int s = 0; s < 4; ++s) acc2[s] = zero8();
    for (int k0 = 0; k0 < 128; k0 += 32) {
        v16bf a = load_frag_a(A2 + (wv << 4) * PH_SA2 + k0, PH_SA2);
        v16bf bfr[4];
        #pragma unroll
        for (int s = 0; s < 4; ++s)
            bfr[s] = load_frag_b(W3s + (s << 4) * PH_SK3 + k0, PH_SK3);
        #pragma unroll
        for (int s = 0; s < 4; ++s)
            acc2[s] = __builtin_amdgcn_wmma_f32_16x16x32_bf16(
                false, a, false, bfr[s], (short)0, acc2[s], false, false);
    }
    __syncthreads();
    // epilogue 2 -> A3 (fp32, reuses A1 region)
    #pragma unroll
    for (int s = 0; s < 4; ++s) {
        int n = (s << 4) + n16;
        float sc = s3[n], sh = t3[n];
        #pragma unroll
        for (int r = 0; r < 8; ++r) {
            int m = (wv << 4) + r + (hi << 3);
            A3[m * PH_SA3 + n] = fmaxf(acc2[s][r] * sc + sh, 0.0f);
        }
    }
    __syncthreads();

    // --- layer 4: 64 -> 1 dot, write score[b, i, j] ---
    if (tid < 128) {
        const float* ar = A3 + tid * PH_SA3;
        float sum = b4[0];
        #pragma unroll
        for (int c = 0; c < 64; ++c) sum += ar[c] * w4s[c];
        int i = i0 + (tid >> 4);
        int j = j0 + (tid & 15);
        out[((size_t)bB << 16) + ((size_t)i << 8) + j] = sum;
    }
}

// ---------------------------------------------------------------------------
// Host side
// ---------------------------------------------------------------------------
static inline void launch_gemm(const float* A, const float* W, const float* bias,
                               float* C, int M, int N, int K, int ep,
                               const float* g, const float* b,
                               const float* m, const float* v,
                               hipStream_t s)
{
    dim3 grid((N + 63) / 64, (M + 127) / 128);
    k_gemm_bf16<<<grid, 256, 0, s>>>(A, W, bias, C, M, N, K, ep, g, b, m, v);
}

// Workspace layout (float offsets)
#define W_CONCAT 0u        // 512*66
#define W_X      40960u    // 512*768   (persistent token features)
#define W_T1     434176u   // 512*2304  (qkv / ffn hidden)
#define W_T2     1613824u  // 512*768   (attn out / mlp hidden)
#define W_T3     2007040u  // 512*768   (proj out)
#define W_XI     2400256u  // 512*256
#define W_XJ     2531328u  // 512*256

extern "C" void kernel_launch(void* const* d_in, const int* in_sizes, int n_in,
                              void* d_out, int out_size, void* d_ws, size_t ws_size,
                              hipStream_t stream) {
    (void)in_sizes; (void)n_in; (void)out_size; (void)ws_size;
    // Input leaf order (setup_inputs dict order):
    // 0 detected_node, 1 visual_descriptor,
    // 2..9   mlp: w1,b1,bn1{g,b,m,v},w2,b2
    // 10+12L enc layer L: w_qkv,b_qkv,w_out,b_out,w1,b1,w2,b2,ln1_g,ln1_b,ln2_g,ln2_b
    // 34+20h head h: w1,b1,bn1{g,b,m,v},w2,b2,bn2{g,b,m,v},w3,b3,bn3{g,b,m,v},w4,b4
    auto F = [&](int i) { return (const float*)d_in[i]; };
    float* ws  = (float*)d_ws;
    float* out = (float*)d_out;
    float* cc = ws + W_CONCAT;
    float* x  = ws + W_X;
    float* t1 = ws + W_T1;
    float* t2 = ws + W_T2;
    float* t3 = ws + W_T3;
    float* xi = ws + W_XI;
    float* xj = ws + W_XJ;

    // Embed MLP: concat -> 256 (bias+bn+relu) -> 768
    k_concat<<<dim3((512 * 66 + 255) / 256), 256, 0, stream>>>(F(1), F(0), cc);
    launch_gemm(cc, F(2), F(3), t2, 512, 256, 66, 2, F(4), F(5), F(6), F(7), stream);
    launch_gemm(t2, F(8), F(9), x, 512, 768, 256, 0, nullptr, nullptr, nullptr, nullptr, stream);

    // Two encoder layers (post-norm, S=2 attention)
    for (int L = 0; L < 2; ++L) {
        int p = 10 + 12 * L;
        launch_gemm(x, F(p + 0), F(p + 1), t1, 512, 2304, 768, 0, nullptr, nullptr, nullptr, nullptr, stream);
        k_attn<<<dim3(12), 256, 0, stream>>>(t1, t2);
        launch_gemm(t2, F(p + 2), F(p + 3), t3, 512, 768, 768, 0, nullptr, nullptr, nullptr, nullptr, stream);
        k_add_ln<<<dim3(512), 256, 0, stream>>>(x, t3, F(p + 8), F(p + 9));
        launch_gemm(x, F(p + 4), F(p + 5), t1, 512, 2048, 768, 1, nullptr, nullptr, nullptr, nullptr, stream);
        launch_gemm(t1, F(p + 6), F(p + 7), t3, 512, 768, 2048, 0, nullptr, nullptr, nullptr, nullptr, stream);
        k_add_ln<<<dim3(512), 256, 0, stream>>>(x, t3, F(p + 10), F(p + 11));
    }

    // Two score heads: thin xi/xj GEMMs, then fused pairwise MLP
    for (int hh = 0; hh < 2; ++hh) {
        int p = 34 + 20 * hh;
        const float* wa = F(p + 0);                 // rows 0..767 of (1536,256)
        const float* wb = F(p + 0) + 768 * 256;     // rows 768..1535
        launch_gemm(x, wa, nullptr, xi, 512, 256, 768, 0, nullptr, nullptr, nullptr, nullptr, stream);
        launch_gemm(x, wb, F(p + 1), xj, 512, 256, 768, 0, nullptr, nullptr, nullptr, nullptr, stream);
        k_pairhead<<<dim3(16, 32, 2), 256, PH_SMEM, stream>>>(
            xi, xj,
            F(p + 2), F(p + 3), F(p + 4), F(p + 5),
            F(p + 6), F(p + 7), F(p + 8), F(p + 9), F(p + 10), F(p + 11),
            F(p + 12), F(p + 13), F(p + 14), F(p + 15), F(p + 16), F(p + 17),
            F(p + 18), F(p + 19),
            out + (size_t)hh * 131072);
    }
}
